// AttentionPairBias_317827580837
// MI455X (gfx1250) — matrix operand order repaired and checked by reading, not compiled, not run on hardware
//
#include <hip/hip_runtime.h>
#include <hip/hip_bf16.h>
#include <math.h>

// ---------------------------------------------------------------------------
// AttentionPairBias for MI455X (gfx1250): wave32 WMMA f32_16x16x32_f16 path.
// B=2, N=768, c_s=768, H=16, hd=48 (padded to 64), c_z=128.
// ---------------------------------------------------------------------------

typedef _Float16 half8  __attribute__((ext_vector_type(8)));
typedef _Float16 half16 __attribute__((ext_vector_type(16)));
typedef float    v8f    __attribute__((ext_vector_type(8)));

#define CAT16(a, b) __builtin_shufflevector(a, b, 0,1,2,3,4,5,6,7,8,9,10,11,12,13,14,15)

#define BATCH 2
#define SEQ   768
#define CS    768
#define HEADS 16
#define HD    48
#define HDP   64
#define CZ    128
#define MROWS (BATCH * SEQ)   // 1536

enum { OUT_F32_RM = 0, OUT_HALF_SIG = 1, OUT_QK = 2, OUT_VT = 3 };

// ---------------------------------------------------------------------------
// fp32 -> fp16 elementwise conversion
// ---------------------------------------------------------------------------
__global__ void f2h_kernel(const float* __restrict__ src, _Float16* __restrict__ dst,
                           unsigned long long n) {
    unsigned long long i = (unsigned long long)blockIdx.x * blockDim.x + threadIdx.x;
    if (i < n) dst[i] = (_Float16)src[i];
}

// ---------------------------------------------------------------------------
// Generic WMMA GEMM: C[m,n] = act(((sum_k A[m,k] * W[n,k]) + cbias[n]) * scale)
// A: M x K half row-major.  W: Nc x K half row-major (i.e. computes A @ W^T).
// One wave computes one 16x16 tile; K-loop in steps of 32.
// ---------------------------------------------------------------------------
__global__ void gemm_wmma(const _Float16* __restrict__ A,
                          const _Float16* __restrict__ W,
                          const float* __restrict__ cbias,  // nullable
                          float scale, int mode,
                          void* __restrict__ dst,
                          int M, int K, int Nc) {
    int wid  = (blockIdx.x * blockDim.x + threadIdx.x) >> 5;
    int lane = threadIdx.x & 31;
    int tilesM = M >> 4;
    int tM = wid % tilesM;
    int tN = wid / tilesM;
    if (tN >= (Nc >> 4)) return;

    int lo   = lane & 15;
    int mrow = tM * 16 + lo;
    int ncol = tN * 16 + lo;
    int offA = (lane < 16) ? 0 : 8;    // A fragment K-octet offset (ISA layout)
    int offB = (lane < 16) ? 0 : 16;   // unused: B fragment uses offA like A

    const _Float16* arow = A + (size_t)mrow * K;
    const _Float16* wrow = W + (size_t)ncol * K;

    v8f acc = {};
    for (int kk = 0; kk < K; kk += 32) {
        half8 a0 = *(const half8*)(arow + kk + offA);
        half8 a1 = *(const half8*)(arow + kk + 16 + offA);
        half8 b0 = *(const half8*)(wrow + kk + offA);
        half8 b1 = *(const half8*)(wrow + kk + 16 + offA);
        acc = __builtin_amdgcn_wmma_f32_16x16x32_f16(
            false, CAT16(a0, a1), false, CAT16(b0, b1), (short)0, acc, false, false);
    }

    float cb = cbias ? cbias[ncol] : 0.0f;
#pragma unroll
    for (int r = 0; r < 8; ++r) {
        int m = tM * 16 + r + ((lane >= 16) ? 8 : 0);
        float val = (acc[r] + cb) * scale;
        if (mode == OUT_F32_RM) {
            ((float*)dst)[(size_t)m * Nc + ncol] = val;
        } else if (mode == OUT_HALF_SIG) {
            float sg = 1.0f / (1.0f + __expf(-val));
            ((_Float16*)dst)[(size_t)m * Nc + ncol] = (_Float16)sg;
        } else if (mode == OUT_QK) {       // (B,H,SEQ,HDP) padded head layout
            int h = ncol / HD, d = ncol % HD;
            int b = m / SEQ,  i = m % SEQ;
            ((_Float16*)dst)[(((size_t)b * HEADS + h) * SEQ + i) * HDP + d] = (_Float16)val;
        } else {                           // OUT_VT: (B,H,HDP,SEQ) transposed V
            int h = ncol / HD, d = ncol % HD;
            int b = m / SEQ,  i = m % SEQ;
            ((_Float16*)dst)[(((size_t)b * HEADS + h) * HDP + d) * SEQ + i] = (_Float16)val;
        }
    }
}

// ---------------------------------------------------------------------------
// Pair bias: bias[b,h,i,j] = sum_c LN(z)[b,i,j,c] * Wz[h,c]  +  (1-mask[b,j])*(-INF)
// One wave handles 16 consecutive (b,i,j) rows (same b,i; j tile of 16).
// LayerNorm via lane-pair split of 128 channels; WMMA against Wz (16x128).
// ---------------------------------------------------------------------------
__global__ void pairbias_wmma(const float* __restrict__ z,
                              const float* __restrict__ ln_g,
                              const float* __restrict__ ln_b,
                              const _Float16* __restrict__ Wz,
                              const float* __restrict__ mask,
                              float* __restrict__ bias) {
    __shared__ __align__(16) _Float16 zn[8][16][CZ];
    int w = threadIdx.x >> 5, lane = threadIdx.x & 31;
    long long tile = (long long)blockIdx.x * 8 + w;
    long long r0 = tile * 16;
    int j0 = (int)(r0 % SEQ);
    int i  = (int)((r0 / SEQ) % SEQ);
    int b  = (int)(r0 / ((long long)SEQ * SEQ));

    int row = lane & 15, hs = lane >> 4;
    const float* zp = z + ((((size_t)b * SEQ + i) * SEQ) + (j0 + row)) * CZ + hs * 64;

    float v[64];
    float s = 0.0f, sq = 0.0f;
#pragma unroll
    for (int c = 0; c < 64; c += 4) {
        float4 f = *(const float4*)(zp + c);
        v[c] = f.x; v[c + 1] = f.y; v[c + 2] = f.z; v[c + 3] = f.w;
        s  += f.x + f.y + f.z + f.w;
        sq += f.x * f.x + f.y * f.y + f.z * f.z + f.w * f.w;
    }
    s  += __shfl_xor(s, 16);
    sq += __shfl_xor(sq, 16);
    float mean = s * (1.0f / CZ);
    float var  = sq * (1.0f / CZ) - mean * mean;
    float rstd = rsqrtf(var + 1e-5f);
#pragma unroll
    for (int c = 0; c < 64; ++c) {
        int cc = hs * 64 + c;
        zn[w][row][cc] = (_Float16)((v[c] - mean) * rstd * ln_g[cc] + ln_b[cc]);
    }
    __syncthreads();

    int lo = lane & 15;
    int offA = (lane < 16) ? 0 : 8;
    int offB = (lane < 16) ? 0 : 16;
    const _Float16* wrow = Wz + (size_t)lo * CZ;
    v8f acc = {};
#pragma unroll
    for (int kk = 0; kk < CZ; kk += 32) {
        half8 a0 = *(const half8*)(&zn[w][lo][kk + offA]);
        half8 a1 = *(const half8*)(&zn[w][lo][kk + 16 + offA]);
        half8 b0 = *(const half8*)(wrow + kk + offA);
        half8 b1 = *(const half8*)(wrow + kk + 16 + offA);
        acc = __builtin_amdgcn_wmma_f32_16x16x32_f16(
            false, CAT16(a0, a1), false, CAT16(b0, b1), (short)0, acc, false, false);
    }

    int h = lo;  // D-fragment column = head
#pragma unroll
    for (int r = 0; r < 8; ++r) {
        int m = r + ((lane >= 16) ? 8 : 0);
        int j = j0 + m;
        float mterm = (1.0f - mask[b * SEQ + j]) * (-1000000.0f);
        bias[((((size_t)b * HEADS + h) * SEQ + i) * SEQ) + j] = acc[r] + mterm;
    }
}

// ---------------------------------------------------------------------------
// Flash attention: one wave per (b, h, 16-query tile). Online softmax on the
// WMMA C-fragment layout; P re-shaped D->A through 512B of LDS per wave.
// q pre-scaled by 1/sqrt(hd); bias pre-includes mask term.
// ---------------------------------------------------------------------------
__global__ void flash_wmma(const _Float16* __restrict__ q,   // (B,H,SEQ,HDP)
                           const _Float16* __restrict__ k,   // (B,H,SEQ,HDP)
                           const _Float16* __restrict__ vt,  // (B,H,HDP,SEQ)
                           const float*    __restrict__ bias,// (B,H,SEQ,SEQ)
                           const _Float16* __restrict__ g,   // (B*SEQ, CS) sigmoid gate
                           _Float16*       __restrict__ og)  // (B*SEQ, CS)
{
    __shared__ __align__(16) _Float16 P[4][16][16];
    int w = threadIdx.x >> 5, lane = threadIdx.x & 31;
    int wid = blockIdx.x * 4 + w;
    int qt = wid % (SEQ / 16);
    int h  = (wid / (SEQ / 16)) % HEADS;
    int b  = wid / ((SEQ / 16) * HEADS);
    int i0 = qt * 16;

    int lo = lane & 15;
    int offA = (lane < 16) ? 0 : 8;
    int offB = (lane < 16) ? 0 : 16;

    const _Float16* qb = q  + (((size_t)b * HEADS + h) * SEQ) * HDP;
    const _Float16* kb = k  + (((size_t)b * HEADS + h) * SEQ) * HDP;
    const _Float16* vb = vt + (((size_t)b * HEADS + h) * HDP) * SEQ;
    const float* biasb = bias + (((size_t)b * HEADS + h) * SEQ + i0) * SEQ;

    // Q A-fragments for K-dim 0..31 and 32..63 (d=48..63 zero-padded in memory)
    const _Float16* qrow = qb + (size_t)(i0 + lo) * HDP;
    half16 q0 = CAT16(*(const half8*)(qrow + offA),      *(const half8*)(qrow + 16 + offA));
    half16 q1 = CAT16(*(const half8*)(qrow + 32 + offA), *(const half8*)(qrow + 48 + offA));

    float mrow[8], lrow[8];
    v8f o0 = {}, o1 = {}, o2 = {};
#pragma unroll
    for (int r = 0; r < 8; ++r) { mrow[r] = -3.0e38f; lrow[r] = 0.0f; }

    half8  zero8  = {0, 0, 0, 0, 0, 0, 0, 0};
    half16 zero16 = CAT16(zero8, zero8);

    for (int j0 = 0; j0 < SEQ; j0 += 16) {
        // K B-fragments: B[d, jn] = K[j0+jn, d] (d octets at offA and 16+offA per lane, same k order as Q)
        const _Float16* krow = kb + (size_t)(j0 + lo) * HDP;
        half16 k0 = CAT16(*(const half8*)(krow + offA),      *(const half8*)(krow + 16 + offA));
        half16 k1 = CAT16(*(const half8*)(krow + 32 + offA), *(const half8*)(krow + 48 + offA));

        v8f sc = {};
        sc = __builtin_amdgcn_wmma_f32_16x16x32_f16(false, q0, false, k0, (short)0, sc, false, false);
        sc = __builtin_amdgcn_wmma_f32_16x16x32_f16(false, q1, false, k1, (short)0, sc, false, false);

#pragma unroll
        for (int r = 0; r < 8; ++r) {
            int m = r + ((lane >= 16) ? 8 : 0);
            float sv = sc[r] + biasb[(size_t)m * SEQ + j0 + lo];
            // row-max across the 16 lanes of the half-wave group
            float mx = sv;
            mx = fmaxf(mx, __shfl_xor(mx, 1));
            mx = fmaxf(mx, __shfl_xor(mx, 2));
            mx = fmaxf(mx, __shfl_xor(mx, 4));
            mx = fmaxf(mx, __shfl_xor(mx, 8));
            float mnew = fmaxf(mrow[r], mx);
            float corr = __expf(mrow[r] - mnew);
            float p    = __expf(sv - mnew);
            float ps = p;
            ps += __shfl_xor(ps, 1);
            ps += __shfl_xor(ps, 2);
            ps += __shfl_xor(ps, 4);
            ps += __shfl_xor(ps, 8);
            lrow[r] = lrow[r] * corr + ps;
            mrow[r] = mnew;
            o0[r] *= corr; o1[r] *= corr; o2[r] *= corr;
            P[w][m][lo] = (_Float16)p;   // D-layout -> LDS
        }
        // same-wave DS ops are in-order; re-read P in A-fragment layout
        half8  pa0 = *(const half8*)(&P[w][lo][offA]);
        half16 pa  = CAT16(pa0, zero8);  // K=16..31 of P is zero

        // V B-fragments from transposed layout: jn octet offA per lane, K=16..31 zero
        const _Float16* v0p = vb + (size_t)(0 * 16 + lo) * SEQ + j0;
        const _Float16* v1p = vb + (size_t)(1 * 16 + lo) * SEQ + j0;
        const _Float16* v2p = vb + (size_t)(2 * 16 + lo) * SEQ + j0;
        half16 vf0 = CAT16(*(const half8*)(v0p + offA), zero8);
        half16 vf1 = CAT16(*(const half8*)(v1p + offA), zero8);
        half16 vf2 = CAT16(*(const half8*)(v2p + offA), zero8);

        o0 = __builtin_amdgcn_wmma_f32_16x16x32_f16(false, pa, false, vf0, (short)0, o0, false, false);
        o1 = __builtin_amdgcn_wmma_f32_16x16x32_f16(false, pa, false, vf1, (short)0, o1, false, false);
        o2 = __builtin_amdgcn_wmma_f32_16x16x32_f16(false, pa, false, vf2, (short)0, o2, false, false);
    }

    // epilogue: normalize, gate, store (d = 0..47 across tiles t=0..2)
#pragma unroll
    for (int r = 0; r < 8; ++r) {
        int m = r + ((lane >= 16) ? 8 : 0);
        int i = i0 + m;
        float inv = 1.0f / lrow[r];
        size_t rowbase = ((size_t)b * SEQ + i) * CS + (size_t)h * HD;
        {
            size_t idx = rowbase + 0 * 16 + lo;
            og[idx] = (_Float16)(o0[r] * inv * (float)g[idx]);
        }
        {
            size_t idx = rowbase + 1 * 16 + lo;
            og[idx] = (_Float16)(o1[r] * inv * (float)g[idx]);
        }
        {
            size_t idx = rowbase + 2 * 16 + lo;
            og[idx] = (_Float16)(o2[r] * inv * (float)g[idx]);
        }
    }
}

// ---------------------------------------------------------------------------
// host launcher
// ---------------------------------------------------------------------------
extern "C" void kernel_launch(void* const* d_in, const int* in_sizes, int n_in,
                              void* d_out, int out_size, void* d_ws, size_t ws_size,
                              hipStream_t stream) {
    (void)in_sizes; (void)n_in; (void)out_size; (void)ws_size;
    const float* s_f    = (const float*)d_in[0];
    const float* z_f    = (const float*)d_in[1];
    const float* mask_f = (const float*)d_in[2];
    const float* kin_f  = (const float*)d_in[3];
    const float* Wq_f   = (const float*)d_in[4];
    const float* bq_f   = (const float*)d_in[5];
    const float* Wk_f   = (const float*)d_in[6];
    const float* Wv_f   = (const float*)d_in[7];
    const float* Wg_f   = (const float*)d_in[8];
    const float* lng_f  = (const float*)d_in[9];
    const float* lnb_f  = (const float*)d_in[10];
    const float* Wz_f   = (const float*)d_in[11];
    const float* Wo_f   = (const float*)d_in[12];
    // d_in[13] = multiplicity (== 1 in setup; jnp.repeat with 1 is identity)
    float* out = (float*)d_out;

    char* ws = (char*)d_ws;
    size_t off = 0;
    auto carve = [&](size_t bytes) -> void* {
        void* p = ws + off;
        off = (off + bytes + 255) & ~(size_t)255;
        return p;
    };

    const size_t nSK = (size_t)MROWS * CS;            // 1,179,648 elements
    const size_t nW  = (size_t)CS * CS;
    _Float16* s_h   = (_Float16*)carve(nSK * 2);
    _Float16* kin_h = (_Float16*)carve(nSK * 2);
    _Float16* Wq_h  = (_Float16*)carve(nW * 2);
    _Float16* Wk_h  = (_Float16*)carve(nW * 2);
    _Float16* Wv_h  = (_Float16*)carve(nW * 2);
    _Float16* Wg_h  = (_Float16*)carve(nW * 2);
    _Float16* Wo_h  = (_Float16*)carve(nW * 2);
    _Float16* Wz_h  = (_Float16*)carve((size_t)HEADS * CZ * 2);
    const size_t nQKV = (size_t)BATCH * HEADS * SEQ * HDP;
    _Float16* q_ws  = (_Float16*)carve(nQKV * 2);
    _Float16* k_ws  = (_Float16*)carve(nQKV * 2);
    _Float16* vT_ws = (_Float16*)carve(nQKV * 2);
    _Float16* g_h   = (_Float16*)carve(nSK * 2);
    _Float16* og_h  = (_Float16*)carve(nSK * 2);
    float* bias_ws  = (float*)carve((size_t)BATCH * HEADS * SEQ * SEQ * 4);

    // zero q/k/v pads (head dim 48..63) — rewritten every call, deterministic
    hipMemsetAsync(q_ws,  0, nQKV * 2, stream);
    hipMemsetAsync(k_ws,  0, nQKV * 2, stream);
    hipMemsetAsync(vT_ws, 0, nQKV * 2, stream);

    // stage 1: f32 -> f16 conversions
    auto conv = [&](const float* src, _Float16* dst, size_t n) {
        unsigned int blocks = (unsigned int)((n + 255) / 256);
        f2h_kernel<<<blocks, 256, 0, stream>>>(src, dst, n);
    };
    conv(s_f,   s_h,   nSK);
    conv(kin_f, kin_h, nSK);
    conv(Wq_f,  Wq_h,  nW);
    conv(Wk_f,  Wk_h,  nW);
    conv(Wv_f,  Wv_h,  nW);
    conv(Wg_f,  Wg_h,  nW);
    conv(Wo_f,  Wo_h,  nW);
    conv(Wz_f,  Wz_h,  (size_t)HEADS * CZ);

    // stage 2: projections via WMMA GEMM (M=1536, K=768, Nc=768)
    const int tiles = (MROWS / 16) * (CS / 16);       // 4608 waves
    const int gemmBlocks = tiles / 8;                 // 8 waves / 256-thread block
    const float qscale = 1.0f / sqrtf((float)HD);
    gemm_wmma<<<gemmBlocks, 256, 0, stream>>>(s_h,   Wq_h, bq_f,    qscale, OUT_QK,
                                              q_ws,  MROWS, CS, CS);
    gemm_wmma<<<gemmBlocks, 256, 0, stream>>>(kin_h, Wk_h, nullptr, 1.0f,   OUT_QK,
                                              k_ws,  MROWS, CS, CS);
    gemm_wmma<<<gemmBlocks, 256, 0, stream>>>(kin_h, Wv_h, nullptr, 1.0f,   OUT_VT,
                                              vT_ws, MROWS, CS, CS);
    gemm_wmma<<<gemmBlocks, 256, 0, stream>>>(s_h,   Wg_h, nullptr, 1.0f,   OUT_HALF_SIG,
                                              g_h,   MROWS, CS, CS);

    // stage 3: pair bias (LayerNorm(z) @ Wz^T + mask fold); 73728 row tiles
    const long long biasTiles = (long long)BATCH * SEQ * SEQ / 16;
    pairbias_wmma<<<(unsigned int)(biasTiles / 8), 256, 0, stream>>>(
        z_f, lng_f, lnb_f, Wz_h, mask_f, bias_ws);

    // stage 4: flash attention + gate; 1536 waves, 4 per block
    const int flashWaves = BATCH * HEADS * (SEQ / 16);
    flash_wmma<<<flashWaves / 4, 128, 0, stream>>>(q_ws, k_ws, vT_ws, bias_ws, g_h, og_h);

    // stage 5: output projection -> fp32 d_out
    gemm_wmma<<<gemmBlocks, 256, 0, stream>>>(og_h, Wo_h, nullptr, 1.0f, OUT_F32_RM,
                                              out, MROWS, CS, CS);
}
